// SingleP_Conformer_3530463117920
// MI455X (gfx1250) — compile-verified, matrix operand order checked by reading, not run on hardware
//
#include <hip/hip_runtime.h>
#include <hip/hip_bf16.h>

// ---------------------------------------------------------------------------
// SingleP Conformer decoder pipeline for MI455X (gfx1250, wave32, WMMA)
// All matrix math routed through v_wmma_f32_16x16x32_f16 (f32 accumulate).
// Branch-free clamped loads, b128 vector loads, LDS-staged tiles, uniform
// fast-path epilogues, float4 softmax/top-k scans.
// ---------------------------------------------------------------------------

typedef __attribute__((ext_vector_type(16))) _Float16 v16h;
typedef __attribute__((ext_vector_type(8)))  float    v8f;

#define DEV_INLINE __device__ __forceinline__

// CDNA5 16-bit A/B matrix VGPR layout (ISA 7.12.2):
// lane = (half<<4)|idx, VGPR v, packed element e ->
//   K = (v>>2)*16 + half*8 + (v&3)*2 + e
// => for fixed lane, elements [0..7] cover K = half*8 + 0..7 (contiguous)
//    and elements [8..15] cover K = 16 + half*8 + 0..7 (contiguous).
DEV_INLINE int kmap(int v, int e, int half) {
  return ((v >> 2) << 4) + (half << 3) + ((v & 3) << 1) + e;
}

DEV_INLINE float sigmoidf_(float x) { return 1.f / (1.f + __expf(-x)); }

// convert 8 contiguous floats (two float4) into packed f16 operand elements
DEV_INLINE void cvt8(v16h& d, int base, float4 a, float4 b, float m) {
  d[base + 0] = (_Float16)(a.x * m);
  d[base + 1] = (_Float16)(a.y * m);
  d[base + 2] = (_Float16)(a.z * m);
  d[base + 3] = (_Float16)(a.w * m);
  d[base + 4] = (_Float16)(b.x * m);
  d[base + 5] = (_Float16)(b.y * m);
  d[base + 6] = (_Float16)(b.z * m);
  d[base + 7] = (_Float16)(b.w * m);
}

// ---------------------------------------------------------------------------
// Generic WMMA GEMM: Y[M,N] = act(X[M,K] @ W[K,N] + bias[N])
// K multiple of 32 (64 or 256 here). 4 waves/block; each wave owns one 16x16
// D tile (block = 64 rows x 16 cols). B tile (32x16) staged once per chunk
// in LDS, shared by all 4 waves.
// ---------------------------------------------------------------------------
#define ACT_NONE 0
#define ACT_RELU 1

__global__ __launch_bounds__(128)
void k_gemm(const float* __restrict__ X, const float* __restrict__ W,
            const float* __restrict__ bias, float* __restrict__ Y,
            int M, int K, int N, int act)
{
  __shared__ __align__(16) float bs[32 * 16];

  const int t    = threadIdx.x;
  const int lane = t & 31;
  const int wave = t >> 5;
  const int m0   = blockIdx.x * 64 + wave * 16;
  const int n0   = blockIdx.y * 16;
  const int half = lane >> 4;
  const int mr   = lane & 15;

  const int  row  = m0 + mr;
  const bool rok  = row < M;
  const int  rowc = rok ? row : 0;         // clamped: always-valid address
  const float am  = rok ? 1.f : 0.f;       // value mask (branch-free)
  const int  n    = n0 + mr;

  const float* Xr = X + (size_t)rowc * K + half * 8;

  // cooperative B-tile load coords: thread t -> (k-row, 4-col group)
  const int bk = t >> 2;                   // 0..31
  const int bn = (t & 3) * 4;              // 0,4,8,12

  v8f acc = {};
  for (int k0 = 0; k0 < K; k0 += 32) {
    *(float4*)&bs[bk * 16 + bn] =
        *(const float4*)&W[(size_t)(k0 + bk) * N + n0 + bn];
    __syncthreads();

    v16h a, b;
    {
      const float4 a0 = *(const float4*)(Xr + k0);
      const float4 a1 = *(const float4*)(Xr + k0 + 4);
      const float4 a2 = *(const float4*)(Xr + k0 + 16);
      const float4 a3 = *(const float4*)(Xr + k0 + 20);
      cvt8(a, 0, a0, a1, am);
      cvt8(a, 8, a2, a3, am);
    }
#pragma unroll
    for (int v = 0; v < 8; ++v) {
#pragma unroll
      for (int e = 0; e < 2; ++e)
        b[2 * v + e] = (_Float16)bs[kmap(v, e, half) * 16 + mr];
    }
    acc = __builtin_amdgcn_wmma_f32_16x16x32_f16(
        false, a, false, b, (short)0, acc, false, false);
    __syncthreads();
  }

  const float bnv = bias ? bias[n] : 0.f;
  const bool relu = (act == ACT_RELU);
  if (m0 + 16 <= M) {                       // uniform fast path: full tile
#pragma unroll
    for (int r = 0; r < 8; ++r) {
      float v = acc[r] + bnv;
      if (relu) v = fmaxf(v, 0.f);
      Y[(size_t)(m0 + half * 8 + r) * N + n] = v;
    }
  } else {                                  // tail tile: guarded stores
#pragma unroll
    for (int r = 0; r < 8; ++r) {
      const int orow = m0 + half * 8 + r;
      if (orow < M) {
        float v = acc[r] + bnv;
        if (relu) v = fmaxf(v, 0.f);
        Y[(size_t)orow * N + n] = v;
      }
    }
  }
}

// ---------------------------------------------------------------------------
// Fused attention: one wave per (b, h, 16-query tile).
//   logits = (Q/sqrt(dk)) @ K^T  -> LDS [16 x 768]
//   softmax (optional top-16 gate, tie-aware threshold), renormalize (float4)
//   O = W @ V via WMMA; V chunks staged through LDS with coalesced b128
// Q/K/V are projected [B,L,C] tensors; head h = columns [h*16, h*16+16).
// ---------------------------------------------------------------------------
#define NH_  4
#define DK_  16
#define NPAD 768

template <bool GATED>
__global__ __launch_bounds__(32)
void k_attn(const float* __restrict__ Qp, const float* __restrict__ Kp,
            const float* __restrict__ Vp, float* __restrict__ Op,
            int Lq, int Lk)
{
  __shared__ __align__(16) float lds[16 * NPAD];
  __shared__ __align__(16) float vt[32 * 16];

  const int lane = threadIdx.x & 31;
  const int half = lane >> 4;
  const int mr   = lane & 15;
  const int b    = blockIdx.x / NH_;
  const int h    = blockIdx.x % NH_;
  const int qt   = blockIdx.y;

  const float* Qb = Qp + (size_t)b * Lq * 64 + h * DK_;
  const float* Kb = Kp + (size_t)b * Lk * 64 + h * DK_;
  const float* Vb = Vp + (size_t)b * Lk * 64 + h * DK_;

  // tail columns [752,768) never written by logits tiles -> poison low
  for (int idx = lane; idx < 16 * 16; idx += 32) {
    lds[(idx >> 4) * NPAD + 752 + (idx & 15)] = -1e30f;
  }

  // --- Q tile: 16 x dk (dk=16 zero-padded to K=32), pre-scaled 1/sqrt(dk) ---
  v16h aQ;
  {
    const int  qrow  = qt * 16 + mr;
    const bool qok   = qrow < Lq;
    const int  qrowc = qok ? qrow : 0;
    const float* Qr  = Qb + (size_t)qrowc * 64 + half * 8;
    const float4 q0  = *(const float4*)Qr;
    const float4 q1  = *(const float4*)(Qr + 4);
    cvt8(aQ, 0, q0, q1, qok ? 0.25f : 0.f);
#pragma unroll
    for (int i = 8; i < 16; ++i) aQ[i] = (_Float16)0.f;  // K = 16..31 pad
  }

  // --- logits: 47 chunks of 16 keys -> LDS ---
  const int nkc = (Lk + 15) / 16;
  for (int kc = 0; kc < nkc; ++kc) {
    const int  key  = kc * 16 + mr;
    const bool kok  = key < Lk;
    const int  keyc = kok ? key : 0;

    v16h bK;
    {
      const float* Kr = Kb + (size_t)keyc * 64 + half * 8;
      const float4 k0 = *(const float4*)Kr;
      const float4 k1 = *(const float4*)(Kr + 4);
      cvt8(bK, 0, k0, k1, kok ? 1.f : 0.f);
#pragma unroll
      for (int i = 8; i < 16; ++i) bK[i] = (_Float16)0.f;
    }
    v8f zero = {};
    v8f tl = __builtin_amdgcn_wmma_f32_16x16x32_f16(
        false, aQ, false, bK, (short)0, zero, false, false);
#pragma unroll
    for (int r = 0; r < 8; ++r) {
      const int m = half * 8 + r;
      lds[m * NPAD + kc * 16 + mr] = kok ? tl[r] : -1e30f;
    }
  }
  __syncthreads();

  // --- per-row softmax (+ optional top-16 gate). 2 lanes per row.
  //     All scans vectorized as float4 over this lane's half of the row. ---
  {
    float4* lrow4 = (float4*)&lds[mr * NPAD];
    const int c4beg = half * (NPAD / 8);       // 96 float4 per half-row
    const int c4end = c4beg + (NPAD / 8);

    float rm = -1e38f;
    for (int c = c4beg; c < c4end; ++c) {
      const float4 tv = lrow4[c];
      rm = fmaxf(rm, fmaxf(fmaxf(tv.x, tv.y), fmaxf(tv.z, tv.w)));
    }
    rm = fmaxf(rm, __shfl_xor(rm, 16, 32));

    float thr = -1e38f;
    if (GATED) {
      float bound = 1e38f;
      int cnt = 0;
      for (int it = 0; it < 16 && cnt < 16; ++it) {
        float lm = -1e38f;
        for (int c = c4beg; c < c4end; ++c) {
          const float4 tv = lrow4[c];
          lm = fmaxf(lm, tv.x < bound ? tv.x : -1e38f);
          lm = fmaxf(lm, tv.y < bound ? tv.y : -1e38f);
          lm = fmaxf(lm, tv.z < bound ? tv.z : -1e38f);
          lm = fmaxf(lm, tv.w < bound ? tv.w : -1e38f);
        }
        lm = fmaxf(lm, __shfl_xor(lm, 16, 32));
        int lc = 0;
        for (int c = c4beg; c < c4end; ++c) {
          const float4 tv = lrow4[c];
          lc += (tv.x == lm) + (tv.y == lm) + (tv.z == lm) + (tv.w == lm);
        }
        lc += __shfl_xor(lc, 16, 32);
        cnt += lc;
        bound = lm;
      }
      thr = bound;   // 16th-largest value (ties counted), kept with >=
    }

    float s = 0.f;
    for (int c = c4beg; c < c4end; ++c) {
      float4 tv = lrow4[c];
      tv.x = (tv.x >= thr) ? __expf(tv.x - rm) : 0.f;
      tv.y = (tv.y >= thr) ? __expf(tv.y - rm) : 0.f;
      tv.z = (tv.z >= thr) ? __expf(tv.z - rm) : 0.f;
      tv.w = (tv.w >= thr) ? __expf(tv.w - rm) : 0.f;
      lrow4[c] = tv;
      s += (tv.x + tv.y) + (tv.z + tv.w);
    }
    s += __shfl_xor(s, 16, 32);
    const float inv = 1.f / s;
    for (int c = c4beg; c < c4end; ++c) {
      float4 tv = lrow4[c];
      tv.x *= inv; tv.y *= inv; tv.z *= inv; tv.w *= inv;
      lrow4[c] = tv;
    }
  }
  __syncthreads();

  // --- O = W @ V : 24 chunks of 32 keys; weights from LDS as b128,
  //     V chunk staged through LDS with coalesced b128 loads ---
  v8f acc = {};
  for (int kc = 0; kc < NPAD / 32; ++kc) {
    // cooperative V tile load: 32 keys x 16 dims = 128 float4, 4 per lane
#pragma unroll
    for (int j = 0; j < 4; ++j) {
      const int f    = lane + 32 * j;        // float4 index 0..127
      const int ky   = f >> 2;               // key-in-chunk 0..31
      const int dg   = (f & 3) * 4;          // dim group 0,4,8,12
      const int key  = kc * 32 + ky;
      const bool kok = key < Lk;
      const int keyc = kok ? key : 0;
      float4 tv = *(const float4*)&Vb[(size_t)keyc * 64 + dg];
      const float mk = kok ? 1.f : 0.f;
      tv.x *= mk; tv.y *= mk; tv.z *= mk; tv.w *= mk;
      *(float4*)&vt[ky * 16 + dg] = tv;
    }
    __syncthreads();

    v16h aW, bV;
    {
      const float* lr = &lds[mr * NPAD + kc * 32 + half * 8];
      const float4 w0 = *(const float4*)lr;
      const float4 w1 = *(const float4*)(lr + 4);
      const float4 w2 = *(const float4*)(lr + 16);
      const float4 w3 = *(const float4*)(lr + 20);
      cvt8(aW, 0, w0, w1, 1.f);
      cvt8(aW, 8, w2, w3, 1.f);
    }
#pragma unroll
    for (int v = 0; v < 8; ++v) {
#pragma unroll
      for (int e = 0; e < 2; ++e)
        bV[2 * v + e] = (_Float16)vt[kmap(v, e, half) * 16 + mr];
    }
    acc = __builtin_amdgcn_wmma_f32_16x16x32_f16(
        false, aW, false, bV, (short)0, acc, false, false);
    __syncthreads();
  }

  float* Ob = Op + ((size_t)b * Lq + qt * 16) * 64 + h * DK_ + mr;
  if (qt * 16 + 16 <= Lq) {                  // uniform fast path
#pragma unroll
    for (int r = 0; r < 8; ++r)
      Ob[(size_t)(half * 8 + r) * 64] = acc[r];
  } else {
#pragma unroll
    for (int r = 0; r < 8; ++r) {
      const int qrow = qt * 16 + half * 8 + r;
      if (qrow < Lq) Ob[(size_t)(half * 8 + r) * 64] = acc[r];
    }
  }
}

// ---------------------------------------------------------------------------
// LayerNorm over C=64 (optional residual): one wave per row, 2 elems/lane.
// ---------------------------------------------------------------------------
__global__ __launch_bounds__(256)
void k_ln(const float* __restrict__ x, const float* __restrict__ res,
          const float* __restrict__ g, const float* __restrict__ bt,
          float* __restrict__ y, int rows)
{
  const int row = blockIdx.x * blockDim.y + threadIdx.y;
  if (row >= rows) return;
  const int lane = threadIdx.x;
  const size_t base = (size_t)row * 64;

  float v0 = x[base + lane], v1 = x[base + lane + 32];
  if (res) { v0 += res[base + lane]; v1 += res[base + lane + 32]; }

  float s = v0 + v1;
#pragma unroll
  for (int o = 16; o >= 1; o >>= 1) s += __shfl_xor(s, o, 32);
  const float mean = s * (1.f / 64.f);
  const float d0 = v0 - mean, d1 = v1 - mean;
  float q = d0 * d0 + d1 * d1;
#pragma unroll
  for (int o = 16; o >= 1; o >>= 1) q += __shfl_xor(q, o, 32);
  const float rstd = rsqrtf(q * (1.f / 64.f) + 1e-6f);

  y[base + lane]      = d0 * rstd * g[lane]      + bt[lane];
  y[base + lane + 32] = d1 * rstd * g[lane + 32] + bt[lane + 32];
}

// sg = sigmoid(a) * b
__global__ void k_sigmul(const float* __restrict__ a, const float* __restrict__ b,
                         float* __restrict__ y, int n)
{
  const int i = blockIdx.x * blockDim.x + threadIdx.x;
  if (i < n) y[i] = sigmoidf_(a[i]) * b[i];
}

// x = qn + sg * sigmoid(awg)
__global__ void k_glu(const float* __restrict__ qn, const float* __restrict__ sg,
                      const float* __restrict__ awg, float* __restrict__ x, int n)
{
  const int i = blockIdx.x * blockDim.x + threadIdx.x;
  if (i < n) x[i] = qn[i] + sg[i] * sigmoidf_(awg[i]);
}

// out[row] = sigmoid(dot(x[row,:], w) + b) ; one wave per row
__global__ __launch_bounds__(256)
void k_head(const float* __restrict__ x, const float* __restrict__ w,
            const float* __restrict__ b, float* __restrict__ out, int rows)
{
  const int row = blockIdx.x * blockDim.y + threadIdx.y;
  if (row >= rows) return;
  const int lane = threadIdx.x;
  const size_t base = (size_t)row * 64;
  float s = x[base + lane] * w[lane] + x[base + lane + 32] * w[lane + 32];
#pragma unroll
  for (int o = 16; o >= 1; o >>= 1) s += __shfl_xor(s, o, 32);
  if (lane == 0) out[row] = sigmoidf_(s + b[0]);
}

// ---------------------------------------------------------------------------
// Host orchestration
// ---------------------------------------------------------------------------
extern "C" void kernel_launch(void* const* d_in, const int* in_sizes, int n_in,
                              void* d_out, int out_size, void* d_ws, size_t ws_size,
                              hipStream_t stream)
{
  (void)in_sizes; (void)n_in; (void)out_size; (void)ws_size;

  constexpr int B = 4, Lq = 3000, Lk = 750, C = 64, F = 256, DEC = 2;
  constexpr int Mq = B * Lq;            // 12000
  constexpr int Mk = B * Lk;            // 3000
  constexpr size_t SB = (size_t)Mq * C; // 768000
  constexpr size_t SE = (size_t)Mk * C; // 192000

  const float* q        = (const float*)d_in[0];
  const float* enc      = (const float*)d_in[1];
  const float* cal_Wq   = (const float*)d_in[2];
  const float* cal_Wk   = (const float*)d_in[3];
  const float* cal_Wv   = (const float*)d_in[4];
  const float* cal_Wo   = (const float*)d_in[5];
  const float* ln1_g    = (const float*)d_in[6];
  const float* ln1_b    = (const float*)d_in[7];
  const float* cal_W1   = (const float*)d_in[8];
  const float* cal_b1   = (const float*)d_in[9];
  const float* cal_W2   = (const float*)d_in[10];
  const float* cal_b2   = (const float*)d_in[11];
  const float* ln2_g    = (const float*)d_in[12];
  const float* ln2_b    = (const float*)d_in[13];
  const float* m_ln_g   = (const float*)d_in[14];
  const float* m_ln_b   = (const float*)d_in[15];
  const float* m_fc1_W  = (const float*)d_in[16];
  const float* m_fc1_b  = (const float*)d_in[17];
  const float* m_fc2_W  = (const float*)d_in[18];
  const float* m_fc2_b  = (const float*)d_in[19];
  const float* m_Wq     = (const float*)d_in[20];
  const float* m_Wk     = (const float*)d_in[21];
  const float* m_Wv     = (const float*)d_in[22];
  const float* m_Wo     = (const float*)d_in[23];
  const float* m_aln_g  = (const float*)d_in[24];
  const float* m_aln_b  = (const float*)d_in[25];
  const float* fc_W     = (const float*)d_in[26];
  const float* fc_b     = (const float*)d_in[27];

  float* ws  = (float*)d_ws;
  float* x   = ws;            // [Mq,C]
  float* qn  = x   + SB;      // [Mq,C]
  float* sg  = qn  + SB;      // [Mq,C]
  float* awg = sg  + SB;      // [Mq,C]
  float* Qp  = awg + SB;      // [Mq,C]
  float* Op  = Qp  + SB;      // [Mq,C]
  float* O2  = Op  + SB;      // [Mq,C]
  float* t2  = O2  + SB;      // [Mq,C]
  float* Kp  = t2  + SB;      // [Mk,C]
  float* Vp  = Kp  + SE;      // [Mk,C]
  float* t1  = Vp  + SE;      // [Mq,F]

  auto gemm = [&](const float* X, const float* W, const float* bias, float* Y,
                  int M, int K, int N, int act) {
    dim3 g((M + 63) / 64, N / 16);
    k_gemm<<<g, dim3(128), 0, stream>>>(X, W, bias, Y, M, K, N, act);
  };
  auto ln = [&](const float* xi, const float* res, const float* g,
                const float* bt, float* y) {
    k_ln<<<dim3((Mq + 7) / 8), dim3(32, 8), 0, stream>>>(xi, res, g, bt, y, Mq);
  };
  const dim3 attn_grid(B * NH_, (Lq + 15) / 16);
  const int NE = Mq * C;
  const dim3 ew_grid((NE + 255) / 256);

  // ---- cross_attn_layer ----
  gemm(q,   cal_Wq, nullptr, Qp, Mq, C, C, ACT_NONE);
  gemm(enc, cal_Wk, nullptr, Kp, Mk, C, C, ACT_NONE);
  gemm(enc, cal_Wv, nullptr, Vp, Mk, C, C, ACT_NONE);
  k_attn<false><<<attn_grid, 32, 0, stream>>>(Qp, Kp, Vp, Op, Lq, Lk);
  gemm(Op, cal_Wo, nullptr, O2, Mq, C, C, ACT_NONE);
  ln(O2, q, ln1_g, ln1_b, x);

  // ---- FFN ----
  gemm(x,  cal_W1, cal_b1, t1, Mq, C, F, ACT_RELU);
  gemm(t1, cal_W2, cal_b2, t2, Mq, F, C, ACT_NONE);
  ln(t2, x, ln2_g, ln2_b, x);

  // ---- MGAN decoder layers ----
  for (int i = 0; i < DEC; ++i) {
    const size_t wOff = (size_t)i * C * C;
    const size_t bOff = (size_t)i * C;
    ln(x, nullptr, m_ln_g + bOff, m_ln_b + bOff, qn);
    gemm(qn, m_fc1_W + wOff, m_fc1_b + bOff, t2, Mq, C, C, ACT_NONE);
    gemm(qn, m_fc2_W + wOff, m_fc2_b + bOff, O2, Mq, C, C, ACT_NONE);
    k_sigmul<<<ew_grid, 256, 0, stream>>>(t2, O2, sg, NE);

    gemm(sg,  m_Wq + wOff, nullptr, Qp, Mq, C, C, ACT_NONE);
    gemm(enc, m_Wk + wOff, nullptr, Kp, Mk, C, C, ACT_NONE);
    gemm(enc, m_Wv + wOff, nullptr, Vp, Mk, C, C, ACT_NONE);
    k_attn<true><<<attn_grid, 32, 0, stream>>>(Qp, Kp, Vp, Op, Lq, Lk);
    gemm(Op, m_Wo + wOff, nullptr, O2, Mq, C, C, ACT_NONE);
    ln(O2, sg, m_aln_g + bOff, m_aln_b + bOff, awg);

    k_glu<<<ew_grid, 256, 0, stream>>>(qn, sg, awg, x, NE);
  }

  // ---- head ----
  k_head<<<dim3((Mq + 7) / 8), dim3(32, 8), 0, stream>>>(x, fc_W, fc_b,
                                                         (float*)d_out, Mq);
}
